// RAFT_20993800143245
// MI455X (gfx1250) — matrix operand order repaired and checked by reading, not compiled, not run on hardware
//
#include <hip/hip_runtime.h>

typedef __attribute__((ext_vector_type(16))) _Float16 v16h;
typedef __attribute__((ext_vector_type(8)))  float    v8f;
typedef unsigned int u32;
typedef __attribute__((ext_vector_type(4))) u32 u32x4;
typedef __attribute__((ext_vector_type(8))) int i32x8;
typedef __attribute__((ext_vector_type(4))) int i32x4;

#if __has_builtin(__builtin_amdgcn_tensor_load_to_lds) && \
    __has_builtin(__builtin_amdgcn_s_wait_tensorcnt)
#define HAVE_TDM 1
#endif

#define ACT_NONE 0
#define ACT_RELU 1
#define ACT_SIG  2
#define ACT_TANH 3

__device__ __forceinline__ float act_apply(float v, int act) {
  switch (act) {
    case ACT_RELU: return fmaxf(v, 0.f);
    case ACT_SIG:  return 1.f / (1.f + __expf(-v));
    case ACT_TANH: return tanhf(v);
  }
  return v;
}

#ifdef HAVE_TDM
// Tensor Data Mover: async-load a 2D f32 tile (rows x cols, row stride in
// elements) from global memory into LDS.  D# layout per CDNA5 ISA 8.3/8.4:
//   group0: count=1 | lds_addr | global_addr | type=2
//   group1: data_size=4B, tensor_dim0/1, tile_dim0=cols, tile_dim1=rows,
//           tensor_dim0_stride = row stride
// This toolchain exposes the 6-arg builtin (groups 2/3 + trailing group all
// zero for a 2D tile, cpol = 0).
__device__ __forceinline__ void tdm_load_tile_f32(
    const float* gsrc, u32 ldsByteOff, int rows, int cols,
    int rowStrideElems, int tdim0, int tdim1)
{
  unsigned long long ga = (unsigned long long)(uintptr_t)gsrc;
  u32x4 g0;
  g0.x = 1u;                                          // count=1 (valid, user)
  g0.y = ldsByteOff;                                  // lds_addr (bytes)
  g0.z = (u32)(ga & 0xffffffffu);                     // global_addr[31:0]
  g0.w = (u32)((ga >> 32) & 0x1ffffffu) | (2u << 30); // addr[56:32] | type=2
  i32x8 g1;
  g1[0] = (2 << 16);                                  // data_size = 4 bytes
  g1[1] = (tdim0 & 0xffff) << 16;                     // tensor_dim0[15:0]
  g1[2] = ((tdim1 & 0xffff) << 16) | ((tdim0 >> 16) & 0xffff);
  g1[3] = (cols << 16) | ((tdim1 >> 16) & 0xffff);    // tile_dim0 | tdim1 hi
  g1[4] = rows & 0xffff;                              // tile_dim1 (tile_dim2=0)
  g1[5] = rowStrideElems;                             // tensor_dim0_stride lo32
  g1[6] = 0;
  g1[7] = 0;
  i32x4 z4 = {0, 0, 0, 0};
  i32x8 z8 = {0, 0, 0, 0, 0, 0, 0, 0};
  __builtin_amdgcn_tensor_load_to_lds(g0, g1, z4, z4, z8, 0);
}
#endif

// ---------------------------------------------------------------------------
// Implicit-GEMM convolution on the WMMA path, LDS-staged.
//   block = 128 threads = 4 waves; all 4 waves share one 16-wide spatial tile
//   and cover 64 output channels (16 each).  Per 32-wide K-step the block
//   cooperatively gathers the 32x16 im2col tile into LDS (double-buffered,
//   one barrier per step); each wave feeds v_wmma_f32_16x16x32_f16 with
//   A = weights (global, L2-resident) and B = LDS tile (ds_load_b128 x2).
// A and B both use contiguous k = 16*(lane/16) + element.  KH/KW are compile-time so the
// k -> (cin,kh,kw) split uses strength-reduced division.  All hot indices are
// 32-bit (no v_mul_u64 in the inner loop).
// ---------------------------------------------------------------------------
template <int KHt, int KWt>
__global__ __launch_bounds__(128) void conv_wmma_t(
    const float* __restrict__ in, const float* __restrict__ wgt,
    const float* __restrict__ bias, float* __restrict__ out,
    int Cin, int Hin, int Win, int Cout,
    int stride, int padH, int padW, int Hout, int Wout,
    int act, float outScale)
{
  constexpr int KK = KHt * KWt;
  const int K   = Cin * KK;
  const int lane = threadIdx.x & 31;
  const int wave = threadIdx.x >> 5;
  const int HWo  = Hout * Wout;
  const int spBase = blockIdx.x * 16;
  const int coBase = blockIdx.y * 64 + wave * 16;
  const int n      = blockIdx.z;
  const bool hi   = lane >= 16;
  const int  am   = lane & 15;   // A row (cout) this lane supplies
  const int  ncol = lane & 15;   // B column (spatial) this lane consumes

  __shared__ alignas(64) _Float16 Bs[2][16][32];   // [buf][spatial][k]

  const float* inN = in + n * (Cin * Hin * Win);

  // ---- cooperative fill coordinates: fixed spatial column per thread ----
  const int fn = threadIdx.x >> 3;        // 0..15: column of the tile
  const int fk = (threadIdx.x & 7) * 4;   // 4 consecutive k per thread
  int fpos = spBase + fn;
  if (fpos >= HWo) fpos = HWo - 1;
  const int foh = fpos / Wout, fow = fpos % Wout;
  const int fihB = foh * stride - padH;
  const int fiwB = fow * stride - padW;

  auto fillB = [&](int buf, int k0) {
    #pragma unroll
    for (int e = 0; e < 4; ++e) {
      const int kb = k0 + fk + e;
      float x = 0.f;
      if (kb < K) {
        const int ci = kb / KK;           // compile-time divisor
        const int r  = kb - ci * KK;
        const int kh = r / KWt;
        const int kw = r - kh * KWt;
        const int ih = fihB + kh;
        const int iw = fiwB + kw;
        if ((unsigned)ih < (unsigned)Hin && (unsigned)iw < (unsigned)Win)
          x = inN[(ci * Hin + ih) * Win + iw];
      }
      Bs[buf][fn][fk + e] = (_Float16)x;
    }
  };

  const bool   mOK  = (coBase + am) < Cout;
  const float* wRow = wgt + (mOK ? (coBase + am) : 0) * K;

  v8f acc = {0.f, 0.f, 0.f, 0.f, 0.f, 0.f, 0.f, 0.f};
  const int nSteps = (K + 31) / 32;
  fillB(0, 0);
  __syncthreads();
  int cur = 0;
  for (int s = 0; s < nSteps; ++s) {
    const int k0 = s * 32;
    if (s + 1 < nSteps) fillB(cur ^ 1, k0 + 32);
    if (mOK) __builtin_prefetch(wRow + k0 + 64, 0, 3);  // global_prefetch path

    // A fragment (weights).  lanes0-15: V0-7 -> K0-15
    //                        lanes16-31: V0-7 -> K16-31
    v16h a;
    #pragma unroll
    for (int v = 0; v < 8; ++v) {
      const int kA = k0 + (hi ? 16 : 0) + (v << 1);
      float w0 = 0.f, w1 = 0.f;
      if (mOK) {
        if (kA     < K) w0 = wRow[kA];
        if (kA + 1 < K) w1 = wRow[kA + 1];
      }
      a[2 * v]     = (_Float16)w0;
      a[2 * v + 1] = (_Float16)w1;
    }
    // B fragment from LDS: lanes0-15 hold K0-15, lanes16-31 K16-31 (32B load)
    const v16h b = *(const v16h*)&Bs[cur][ncol][hi ? 16 : 0];

    acc = __builtin_amdgcn_wmma_f32_16x16x32_f16(false, a, false, b,
                                                 (short)0, acc, false, false);
    __syncthreads();
    cur ^= 1;
  }

  if (spBase + ncol >= HWo) return;
  float* outN = out + n * (Cout * HWo);
  const int mOffs = hi ? 8 : 0;
  #pragma unroll
  for (int r = 0; r < 8; ++r) {
    const int m = coBase + r + mOffs;
    if (m < Cout) {
      float v = (acc[r] + bias[m]) * outScale;
      outN[m * HWo + spBase + ncol] = act_apply(v, act);
    }
  }
}

// ---------------------------------------------------------------------------
// All-pairs correlation: corr[b,m,n] = (1/16) * sum_k f1[b,k,m]*f2[b,k,n]
// 4 waves per block share one B tile (one n-tile, four m-tiles).  The B tile
// (32 rows x 16 contiguous f32, row stride HW) is streamed into LDS by the
// Tensor Data Mover (wave 0 issues, TENSORcnt + barrier sync, double-buffer);
// fallback is a cooperative vector fill when the TDM builtin is unavailable.
// ---------------------------------------------------------------------------
__global__ __launch_bounds__(128) void corr_gemm_kernel(
    const float* __restrict__ f1, const float* __restrict__ f2,
    float* __restrict__ corr, int Dc, int HW, float scale)
{
  const int lane = threadIdx.x & 31;
  const bool hi  = lane >= 16;
  const int wave = threadIdx.x >> 5;
  const int b     = blockIdx.z;
  const int mBase = blockIdx.x * 64 + wave * 16;
  const int nBase = blockIdx.y * 16;
  const float* A  = f1 + b * (Dc * HW);
  const float* Bm = f2 + b * (Dc * HW);
  const int am = mBase + (lane & 15);

  __shared__ alignas(16) float BsT[2][32][16];   // [buf][k][n] (f32, TDM order)

  const int nSteps = Dc / 32;
  int cur = 0;

#ifdef HAVE_TDM
  const bool issuer = (threadIdx.x < 32);        // wave-uniform branch
  const u32 lds0 = (u32)(uintptr_t)&BsT[0][0][0];
  const u32 lds1 = (u32)(uintptr_t)&BsT[1][0][0];
  if (issuer) {
    tdm_load_tile_f32(Bm + nBase, lds0, 32, 16, HW, HW, Dc);
    __builtin_amdgcn_s_wait_tensorcnt(0);
  }
#else
  const int fn = threadIdx.x & 15;               // column
  const int fr = (threadIdx.x >> 4) * 4;         // 4 rows per thread
  auto fillB = [&](int buf, int k0) {
    #pragma unroll
    for (int e = 0; e < 4; ++e)
      BsT[buf][fr + e][fn] = Bm[(k0 + fr + e) * HW + nBase + fn];
  };
  fillB(0, 0);
#endif
  __syncthreads();

  v8f acc = {0.f, 0.f, 0.f, 0.f, 0.f, 0.f, 0.f, 0.f};
  for (int s = 0; s < nSteps; ++s) {
    const int k0 = s * 32;
#ifdef HAVE_TDM
    if (issuer && s + 1 < nSteps)
      tdm_load_tile_f32(Bm + (k0 + 32) * HW + nBase, (cur ? lds0 : lds1),
                        32, 16, HW, HW, Dc);
#else
    if (s + 1 < nSteps) fillB(cur ^ 1, k0 + 32);
#endif
    v16h a;
    #pragma unroll
    for (int v = 0; v < 8; ++v) {
      const int kA = k0 + (hi ? 16 : 0) + (v << 1);
      a[2 * v]     = (_Float16)A[kA * HW + am];
      a[2 * v + 1] = (_Float16)A[(kA + 1) * HW + am];
    }
    const int kbOff = hi ? 16 : 0;
    v16h bf;
    #pragma unroll
    for (int e = 0; e < 16; ++e)
      bf[e] = (_Float16)BsT[cur][kbOff + e][lane & 15];

    acc = __builtin_amdgcn_wmma_f32_16x16x32_f16(false, a, false, bf,
                                                 (short)0, acc, false, false);
#ifdef HAVE_TDM
    if (issuer) __builtin_amdgcn_s_wait_tensorcnt(0);
#endif
    __syncthreads();
    cur ^= 1;
  }
  const int mOffs = hi ? 8 : 0;
  #pragma unroll
  for (int r = 0; r < 8; ++r) {
    const int m = mBase + r + mOffs;
    corr[(b * HW + m) * HW + nBase + (lane & 15)] = acc[r] * scale;
  }
}

// ---------------------------------------------------------------------------
// Scalar conv for tiny Cout (fh2: 2 channels).
// ---------------------------------------------------------------------------
__global__ void conv_scalar_kernel(
    const float* __restrict__ in, const float* __restrict__ wgt,
    const float* __restrict__ bias, float* __restrict__ out,
    int Nb, int Cin, int Hin, int Win, int Cout, int KH, int KW,
    int st, int pH, int pW, int Hout, int Wout, int act, float scale)
{
  const int HWo = Hout * Wout;
  const int idx = blockIdx.x * blockDim.x + threadIdx.x;
  const int total = Nb * Cout * HWo;
  if (idx >= total) return;
  const int pos = idx % HWo;
  int t = idx / HWo;
  const int co = t % Cout;
  const int n  = t / Cout;
  const int oh = pos / Wout, ow = pos % Wout;
  const float* wp = wgt + co * (Cin * KH * KW);
  const float* ip = in + n * (Cin * Hin * Win);
  float acc = bias[co];
  for (int ci = 0; ci < Cin; ++ci) {
    const float* ipc = ip + ci * (Hin * Win);
    for (int kh = 0; kh < KH; ++kh) {
      const int ih = oh * st - pH + kh;
      if ((unsigned)ih >= (unsigned)Hin) continue;
      for (int kw = 0; kw < KW; ++kw) {
        const int iw = ow * st - pW + kw;
        if ((unsigned)iw < (unsigned)Win)
          acc += ipc[ih * Win + iw] * wp[(ci * KH + kh) * KW + kw];
      }
    }
  }
  out[idx] = act_apply(acc * scale, act);
}

// ---------------------------------------------------------------------------
// Instance / batch norm (population var, eps 1e-5), optional affine + relu.
// inst: one block per (n,c) over HW. batch: one block per c over Nb*HW.
// In-place.
// ---------------------------------------------------------------------------
__global__ __launch_bounds__(256) void norm_kernel(
    float* __restrict__ x, const float* __restrict__ g,
    const float* __restrict__ b, int C, int HW, int Nb, int inst, int relu)
{
  __shared__ float s1[256], s2[256];
  const int t = threadIdx.x;
  int c, n0, nCount;
  if (inst) { const int bc = blockIdx.x; c = bc % C; n0 = bc / C; nCount = 1; }
  else      { c = blockIdx.x; n0 = 0; nCount = Nb; }
  const int count = nCount * HW;
  float sum = 0.f, sq = 0.f;
  for (int i = t; i < count; i += 256) {
    const int n = n0 + i / HW;
    const int o = i % HW;
    const float v = x[(n * C + c) * HW + o];
    sum += v; sq += v * v;
  }
  s1[t] = sum; s2[t] = sq; __syncthreads();
  for (int s = 128; s > 0; s >>= 1) {
    if (t < s) { s1[t] += s1[t + s]; s2[t] += s2[t + s]; }
    __syncthreads();
  }
  const float mean = s1[0] / (float)count;
  const float var  = s2[0] / (float)count - mean * mean;
  const float rstd = rsqrtf(var + 1e-5f);
  const float gg = g ? g[c] : 1.f;
  const float bb = b ? b[c] : 0.f;
  for (int i = t; i < count; i += 256) {
    const int n = n0 + i / HW;
    const int o = i % HW;
    const int id = (n * C + c) * HW + o;
    float v = (x[id] - mean) * rstd * gg + bb;
    if (relu) v = fmaxf(v, 0.f);
    x[id] = v;
  }
}

// ------------------------------ elementwise --------------------------------
__global__ void addrelu_kernel(const float* a, const float* b, float* o, int n) {
  int i = blockIdx.x * blockDim.x + threadIdx.x;
  if (i < n) o[i] = fmaxf(a[i] + b[i], 0.f);
}
__global__ void chan_copy_act_kernel(const float* src, float* dst,
    int Cs, int cs0, int Cd, int cd0, int C, int HW, int Nb, int act) {
  int idx = blockIdx.x * blockDim.x + threadIdx.x;
  int total = Nb * C * HW;
  if (idx >= total) return;
  int i = idx % HW; int t = idx / HW; int c = t % C; int nb = t / C;
  dst[(nb * Cd + cd0 + c) * HW + i] =
      act_apply(src[(nb * Cs + cs0 + c) * HW + i], act);
}
__global__ void mul_kernel(const float* a, const float* b, float* o, int n) {
  int i = blockIdx.x * blockDim.x + threadIdx.x;
  if (i < n) o[i] = a[i] * b[i];
}
__global__ void gru_mix_kernel(float* net, const float* z, const float* q, int n) {
  int i = blockIdx.x * blockDim.x + threadIdx.x;
  if (i < n) net[i] = (1.f - z[i]) * net[i] + z[i] * q[i];
}
__global__ void add_ip_kernel(float* a, const float* b, int n) {
  int i = blockIdx.x * blockDim.x + threadIdx.x;
  if (i < n) a[i] += b[i];
}
__global__ void coords_init_kernel(float* c, int B, int H, int W) {
  int HW = H * W;
  int idx = blockIdx.x * blockDim.x + threadIdx.x;
  if (idx >= B * 2 * HW) return;
  int hw = idx % HW; int t = idx / HW; int ch = t % 2;
  c[idx] = (ch == 0) ? (float)(hw % W) : (float)(hw / W);
}
__global__ void flow_kernel(const float* coords, float* flow, int B, int H, int W) {
  int HW = H * W;
  int idx = blockIdx.x * blockDim.x + threadIdx.x;
  if (idx >= B * 2 * HW) return;
  int hw = idx % HW; int t = idx / HW; int ch = t % 2;
  float base = (ch == 0) ? (float)(hw % W) : (float)(hw / W);
  flow[idx] = coords[idx] - base;
}
__global__ void avgpool2_kernel(const float* in, float* out, int Nc, int h, int w) {
  int ho = h / 2, wo = w / 2;
  int idx = blockIdx.x * blockDim.x + threadIdx.x;
  if (idx >= Nc * ho * wo) return;
  int x = idx % wo; int t = idx / wo; int y = t % ho; int n = t / ho;
  const float* p = in + (n * h + 2 * y) * w + 2 * x;
  out[idx] = 0.25f * (p[0] + p[1] + p[w] + p[w + 1]);
}

// Bilinear correlation-pyramid lookup: out (B, 4*81, H, W)
__global__ void corr_lookup_kernel(const float* __restrict__ coords,
    const float* __restrict__ p0, const float* __restrict__ p1,
    const float* __restrict__ p2, const float* __restrict__ p3,
    float* __restrict__ out, int B, int H, int W)
{
  const int HW = H * W;
  int idx = blockIdx.x * blockDim.x + threadIdx.x;
  const int total = B * 324 * HW;
  if (idx >= total) return;
  const int hw = idx % HW; int t = idx / HW;
  const int ch = t % 324; const int b = t / 324;
  const int lvl = ch / 81, q = ch % 81, di = q / 9, dj = q % 9;
  const float cx = coords[(b * 2 + 0) * HW + hw];
  const float cy = coords[(b * 2 + 1) * HW + hw];
  const float s = (float)(1 << lvl);
  const float px = cx / s + (float)(di - 4);
  const float py = cy / s + (float)(dj - 4);
  const float* lv = (lvl == 0) ? p0 : (lvl == 1) ? p1 : (lvl == 2) ? p2 : p3;
  const int hl = H >> lvl, wl = W >> lvl;
  const size_t rec = ((size_t)b * HW + hw) * (hl * wl);
  const float x0f = floorf(px), y0f = floorf(py);
  const float wx = px - x0f, wy = py - y0f;
  const int x0 = (int)x0f, y0 = (int)y0f;
  auto g = [&](int xi, int yi) -> float {
    if (xi < 0 || xi > wl - 1 || yi < 0 || yi > hl - 1) return 0.f;
    return lv[rec + yi * wl + xi];
  };
  const float v00 = g(x0, y0), v10 = g(x0 + 1, y0);
  const float v01 = g(x0, y0 + 1), v11 = g(x0 + 1, y0 + 1);
  out[(b * 324 + ch) * HW + hw] =
      v00 * (1 - wx) * (1 - wy) + v10 * wx * (1 - wy) +
      v01 * (1 - wx) * wy + v11 * wx * wy;
}

// Convex upsampling: out (B,2,8H,8W) from flow (B,2,H,W) and mask (B,576,H,W)
__global__ void upsample_kernel(const float* __restrict__ flow,
    const float* __restrict__ mask, float* __restrict__ out, int B, int H, int W)
{
  const int HW = H * W;
  int idx = blockIdx.x * blockDim.x + threadIdx.x;
  if (idx >= B * 64 * HW) return;
  const int hw = idx % HW; int t = idx / HW;
  const int ij = t % 64; const int b = t / 64;
  const int i = ij / 8, j = ij % 8;
  const int h = hw / W, w = hw % W;
  float m[9], mx = -1e30f;
  #pragma unroll
  for (int k = 0; k < 9; ++k) {
    m[k] = mask[(b * 576 + (k * 64 + ij)) * HW + hw];
    mx = fmaxf(mx, m[k]);
  }
  float se = 0.f;
  #pragma unroll
  for (int k = 0; k < 9; ++k) { m[k] = __expf(m[k] - mx); se += m[k]; }
  const float inv = 1.f / se;
  const int HO = H * 8, WO = W * 8;
  #pragma unroll
  for (int c = 0; c < 2; ++c) {
    float acc = 0.f;
    #pragma unroll
    for (int k = 0; k < 9; ++k) {
      const int hh = h + k / 3 - 1, ww = w + k % 3 - 1;
      float fv = 0.f;
      if (hh >= 0 && hh < H && ww >= 0 && ww < W)
        fv = 8.f * flow[(b * 2 + c) * HW + hh * W + ww];
      acc += m[k] * inv * fv;
    }
    out[((size_t)(b * 2 + c) * HO + (h * 8 + i)) * WO + (w * 8 + j)] = acc;
  }
}

// ===========================================================================
extern "C" void kernel_launch(void* const* d_in, const int* in_sizes, int n_in,
                              void* d_out, int out_size, void* d_ws, size_t ws_size,
                              hipStream_t stream)
{
  (void)in_sizes; (void)n_in; (void)out_size; (void)ws_size;
  const float* image1 = (const float*)d_in[0];
  const float* image2 = (const float*)d_in[1];
  // d_in[2] = iters (device scalar; fixed at the setup value, graph-safe)
  const int ITERS = 12;

  // -- parameter unpacking: JAX pytree leaf order (sorted dict keys) --------
  int ip = 3;
  auto nextA = [&]() -> const float* { return (const float*)d_in[ip++]; };
  struct CP { const float* b; const float* w; };
  struct NP { const float* b; const float* g; };
  auto nextC = [&]() { CP c; c.b = nextA(); c.w = nextA(); return c; };
  auto nextN = [&]() { NP n; n.b = nextA(); n.g = nextA(); return n; };
  struct Blk { CP c1, c2, down; NP n1, n2, n3; bool hasDown; };
  struct Enc { Blk blk[6]; CP conv1, conv2; NP n1; bool batch; };
  const int bstr[6] = {1, 1, 2, 1, 2, 1};
  const int bcin[6] = {64, 64, 64, 96, 96, 128};
  const int bcout[6] = {64, 64, 96, 96, 128, 128};
  auto readEnc = [&](bool batch) {
    Enc e; e.batch = batch;
    for (int i = 0; i < 6; ++i) {
      Blk& b = e.blk[i];
      b.c1 = nextC(); b.c2 = nextC();
      b.hasDown = (bstr[i] == 2);
      if (b.hasDown) b.down = nextC();
      if (batch) { b.n1 = nextN(); b.n2 = nextN(); if (b.hasDown) b.n3 = nextN(); }
    }
    e.conv1 = nextC(); e.conv2 = nextC();
    if (batch) e.n1 = nextN();
    return e;
  };
  Enc cnetP = readEnc(true);   // 'cnet' < 'fnet' < 'update'
  Enc fnetP = readEnc(false);
  CP convc1 = nextC(), convc2 = nextC(), convf1 = nextC(), convf2 = nextC();
  CP convme = nextC();
  CP convq1 = nextC(), convq2 = nextC(), convr1 = nextC(), convr2 = nextC();
  CP convz1 = nextC(), convz2 = nextC();
  CP fh1 = nextC(), fh2 = nextC(), mask1 = nextC(), mask2 = nextC();

  // -- workspace ------------------------------------------------------------
  const int B = 2, H = 48, W = 80, HW = H * W, Dc = 256;
  const int H0 = 384, W0 = 640;
  char* base = (char*)d_ws; size_t off = 0;
  auto alloc = [&](size_t nf) -> float* {
    float* p = (float*)(base + off);
    off += (nf * sizeof(float) + 255) & ~(size_t)255;
    return p;
  };
  const size_t encBuf = (size_t)2 * 64 * 192 * 320;
  float* sA = alloc(encBuf); float* sB = alloc(encBuf);
  float* sC = alloc(encBuf); float* sD = alloc(encBuf);
  float* fmaps   = alloc((size_t)4 * Dc * HW);
  float* cnetOut = alloc((size_t)2 * Dc * HW);
  float* corr = alloc((size_t)B * HW * HW);
  float* pyr1 = alloc((size_t)B * HW * 24 * 40);
  float* pyr2 = alloc((size_t)B * HW * 12 * 20);
  float* pyr3 = alloc((size_t)B * HW * 6 * 10);
  float* net = alloc((size_t)B * 128 * HW);
  float* inp = alloc((size_t)B * 128 * HW);
  float* coords1 = alloc((size_t)B * 2 * HW);
  float* corrL = alloc((size_t)B * 324 * HW);
  float* flowB = alloc((size_t)B * 2 * HW);
  float* c1b = alloc((size_t)B * 256 * HW);
  float* c2b = alloc((size_t)B * 192 * HW);
  float* f1b = alloc((size_t)B * 128 * HW);
  float* f2b = alloc((size_t)B * 64 * HW);
  float* cfb = alloc((size_t)B * 256 * HW);
  float* mfb = alloc((size_t)B * 126 * HW);
  float* xB  = alloc((size_t)B * 256 * HW);
  float* hx  = alloc((size_t)B * 384 * HW);
  float* zB  = alloc((size_t)B * 128 * HW);
  float* rB  = alloc((size_t)B * 128 * HW);
  float* qB  = alloc((size_t)B * 128 * HW);
  float* rn  = alloc((size_t)B * 128 * HW);
  float* t256 = alloc((size_t)B * 256 * HW);
  float* deltaB = alloc((size_t)B * 2 * HW);
  float* maskB  = alloc((size_t)B * 576 * HW);
  float* flow2  = alloc((size_t)B * 2 * HW);

  // -- launchers ------------------------------------------------------------
  auto convW = [&](const float* in, float* out, CP cv, int Nb, int Cin,
                   int Hin, int Win, int Cout, int KH, int KW, int st,
                   int pH, int pW, int act, float scale) {
    const int Hout = (Hin + 2 * pH - KH) / st + 1;
    const int Wout = (Win + 2 * pW - KW) / st + 1;
    dim3 g((Hout * Wout + 15) / 16, (Cout + 63) / 64, Nb);
    dim3 blk(128);
    if (KH == 1 && KW == 1)
      conv_wmma_t<1, 1><<<g, blk, 0, stream>>>(in, cv.w, cv.b, out,
          Cin, Hin, Win, Cout, st, pH, pW, Hout, Wout, act, scale);
    else if (KH == 3 && KW == 3)
      conv_wmma_t<3, 3><<<g, blk, 0, stream>>>(in, cv.w, cv.b, out,
          Cin, Hin, Win, Cout, st, pH, pW, Hout, Wout, act, scale);
    else if (KH == 7 && KW == 7)
      conv_wmma_t<7, 7><<<g, blk, 0, stream>>>(in, cv.w, cv.b, out,
          Cin, Hin, Win, Cout, st, pH, pW, Hout, Wout, act, scale);
    else if (KH == 1 && KW == 5)
      conv_wmma_t<1, 5><<<g, blk, 0, stream>>>(in, cv.w, cv.b, out,
          Cin, Hin, Win, Cout, st, pH, pW, Hout, Wout, act, scale);
    else if (KH == 5 && KW == 1)
      conv_wmma_t<5, 1><<<g, blk, 0, stream>>>(in, cv.w, cv.b, out,
          Cin, Hin, Win, Cout, st, pH, pW, Hout, Wout, act, scale);
  };
  auto convSc = [&](const float* in, float* out, CP cv, int Nb, int Cin,
                    int Hin, int Win, int Cout, int KH, int KW, int st,
                    int pH, int pW, int act, float scale) {
    const int Hout = (Hin + 2 * pH - KH) / st + 1;
    const int Wout = (Win + 2 * pW - KW) / st + 1;
    const int total = Nb * Cout * Hout * Wout;
    conv_scalar_kernel<<<dim3((total + 255) / 256), dim3(256), 0, stream>>>(
        in, cv.w, cv.b, out, Nb, Cin, Hin, Win, Cout, KH, KW, st, pH, pW,
        Hout, Wout, act, scale);
  };
  auto normL = [&](float* x, const NP* np, int Nb, int C, int HWn,
                   bool inst, int relu) {
    const int blocks = inst ? Nb * C : C;
    norm_kernel<<<dim3(blocks), dim3(256), 0, stream>>>(
        x, np ? np->g : nullptr, np ? np->b : nullptr, C, HWn, Nb,
        inst ? 1 : 0, relu);
  };
  auto ccopy = [&](const float* src, float* dst, int Cs, int cs0, int Cd,
                   int cd0, int C, int act) {
    const int total = B * C * HW;
    chan_copy_act_kernel<<<dim3((total + 255) / 256), dim3(256), 0, stream>>>(
        src, dst, Cs, cs0, Cd, cd0, C, HW, B, act);
  };
  auto ewg = [](int n) { return dim3((n + 255) / 256); };

  // -- encoders -------------------------------------------------------------
  auto runEnc = [&](const Enc& e, const float* img, float* out256) {
    const bool batch = e.batch, inst = !batch;
    const int Nb = 2;
    convW(img, sA, e.conv1, Nb, 3, H0, W0, 64, 7, 7, 2, 3, 3, ACT_NONE, 1.f);
    normL(sA, batch ? &e.n1 : nullptr, Nb, 64, 192 * 320, inst, 1);
    float* x = sA; float* y = sB;
    int Hc = 192, Wc = 320;
    for (int i = 0; i < 6; ++i) {
      const Blk& bl = e.blk[i];
      const int cin = bcin[i], cout = bcout[i], s = bstr[i];
      const int Ho = (Hc - 1) / s + 1, Wo = (Wc - 1) / s + 1;
      convW(x, y, bl.c1, Nb, cin, Hc, Wc, cout, 3, 3, s, 1, 1, ACT_NONE, 1.f);
      normL(y, batch ? &bl.n1 : nullptr, Nb, cout, Ho * Wo, inst, 1);
      convW(y, sC, bl.c2, Nb, cout, Ho, Wo, cout, 3, 3, 1, 1, 1, ACT_NONE, 1.f);
      normL(sC, batch ? &bl.n2 : nullptr, Nb, cout, Ho * Wo, inst, 1);
      const float* res = x;
      if (s != 1) {
        convW(x, sD, bl.down, Nb, cin, Hc, Wc, cout, 1, 1, s, 0, 0, ACT_NONE, 1.f);
        normL(sD, batch ? &bl.n3 : nullptr, Nb, cout, Ho * Wo, inst, 0);
        res = sD;
      }
      const int n = Nb * cout * Ho * Wo;
      addrelu_kernel<<<ewg(n), dim3(256), 0, stream>>>(res, sC, y, n);
      float* tmp = x; x = y; y = tmp;
      Hc = Ho; Wc = Wo;
    }
    convW(x, out256, e.conv2, Nb, 128, Hc, Wc, 256, 1, 1, 1, 0, 0, ACT_NONE, 1.f);
  };

  runEnc(fnetP, image1, fmaps);                       // fmap1 (instance norm)
  runEnc(fnetP, image2, fmaps + (size_t)2 * Dc * HW); // fmap2
  runEnc(cnetP, image1, cnetOut);                     // context (batch norm)

  // -- correlation volume + pyramid -----------------------------------------
  {
    dim3 g(HW / 64, HW / 16, B);   // 4 m-tiles per block
    corr_gemm_kernel<<<g, dim3(128), 0, stream>>>(
        fmaps, fmaps + (size_t)2 * Dc * HW, corr, Dc, HW, 1.f / 16.f);
  }
  {
    int Nc = B * HW;
    int n1 = Nc * 24 * 40, n2 = Nc * 12 * 20, n3 = Nc * 6 * 10;
    avgpool2_kernel<<<ewg(n1), dim3(256), 0, stream>>>(corr, pyr1, Nc, 48, 80);
    avgpool2_kernel<<<ewg(n2), dim3(256), 0, stream>>>(pyr1, pyr2, Nc, 24, 40);
    avgpool2_kernel<<<ewg(n3), dim3(256), 0, stream>>>(pyr2, pyr3, Nc, 12, 20);
  }

  // -- context split: net = tanh(cnet[:, :128]), inp = relu(cnet[:, 128:]) --
  ccopy(cnetOut, net, 256, 0, 128, 0, 128, ACT_TANH);
  ccopy(cnetOut, inp, 256, 128, 128, 0, 128, ACT_RELU);
  coords_init_kernel<<<ewg(B * 2 * HW), dim3(256), 0, stream>>>(coords1, B, H, W);

  const int nGate = B * 128 * HW;
  float* outF = (float*)d_out;
  const size_t outStride = (size_t)B * 2 * (8 * H) * (8 * W);

  for (int it = 0; it < ITERS; ++it) {
    flow_kernel<<<ewg(B * 2 * HW), dim3(256), 0, stream>>>(coords1, flowB, B, H, W);
    corr_lookup_kernel<<<ewg(B * 324 * HW), dim3(256), 0, stream>>>(
        coords1, corr, pyr1, pyr2, pyr3, corrL, B, H, W);

    // motion encoder
    convW(corrL, c1b, convc1, B, 324, H, W, 256, 1, 1, 1, 0, 0, ACT_RELU, 1.f);
    convW(c1b, c2b, convc2, B, 256, H, W, 192, 3, 3, 1, 1, 1, ACT_RELU, 1.f);
    convW(flowB, f1b, convf1, B, 2, H, W, 128, 7, 7, 1, 3, 3, ACT_RELU, 1.f);
    convW(f1b, f2b, convf2, B, 128, H, W, 64, 3, 3, 1, 1, 1, ACT_RELU, 1.f);
    ccopy(c2b, cfb, 192, 0, 256, 0, 192, ACT_NONE);
    ccopy(f2b, cfb, 64, 0, 256, 192, 64, ACT_NONE);
    convW(cfb, mfb, convme, B, 256, H, W, 126, 3, 3, 1, 1, 1, ACT_RELU, 1.f);

    // x = [inp | mf | flow], hx = [net | x]
    ccopy(inp, xB, 128, 0, 256, 0, 128, ACT_NONE);
    ccopy(mfb, xB, 126, 0, 256, 128, 126, ACT_NONE);
    ccopy(flowB, xB, 2, 0, 256, 254, 2, ACT_NONE);
    ccopy(net, hx, 128, 0, 384, 0, 128, ACT_NONE);
    ccopy(xB, hx, 256, 0, 384, 128, 256, ACT_NONE);

    // GRU, horizontal (1x5)
    convW(hx, zB, convz1, B, 384, H, W, 128, 1, 5, 1, 0, 2, ACT_SIG, 1.f);
    convW(hx, rB, convr1, B, 384, H, W, 128, 1, 5, 1, 0, 2, ACT_SIG, 1.f);
    mul_kernel<<<ewg(nGate), dim3(256), 0, stream>>>(rB, net, rn, nGate);
    ccopy(rn, hx, 128, 0, 384, 0, 128, ACT_NONE);
    convW(hx, qB, convq1, B, 384, H, W, 128, 1, 5, 1, 0, 2, ACT_TANH, 1.f);
    gru_mix_kernel<<<ewg(nGate), dim3(256), 0, stream>>>(net, zB, qB, nGate);

    // GRU, vertical (5x1)
    ccopy(net, hx, 128, 0, 384, 0, 128, ACT_NONE);
    convW(hx, zB, convz2, B, 384, H, W, 128, 5, 1, 1, 2, 0, ACT_SIG, 1.f);
    convW(hx, rB, convr2, B, 384, H, W, 128, 5, 1, 1, 2, 0, ACT_SIG, 1.f);
    mul_kernel<<<ewg(nGate), dim3(256), 0, stream>>>(rB, net, rn, nGate);
    ccopy(rn, hx, 128, 0, 384, 0, 128, ACT_NONE);
    convW(hx, qB, convq2, B, 384, H, W, 128, 5, 1, 1, 2, 0, ACT_TANH, 1.f);
    gru_mix_kernel<<<ewg(nGate), dim3(256), 0, stream>>>(net, zB, qB, nGate);

    // flow head + mask head
    convW(net, t256, fh1, B, 128, H, W, 256, 3, 3, 1, 1, 1, ACT_RELU, 1.f);
    convSc(t256, deltaB, fh2, B, 256, H, W, 2, 3, 3, 1, 1, 1, ACT_NONE, 1.f);
    convW(net, t256, mask1, B, 128, H, W, 256, 3, 3, 1, 1, 1, ACT_RELU, 1.f);
    convW(t256, maskB, mask2, B, 256, H, W, 576, 1, 1, 1, 0, 0, ACT_NONE, 0.25f);

    add_ip_kernel<<<ewg(B * 2 * HW), dim3(256), 0, stream>>>(coords1, deltaB, B * 2 * HW);
    flow_kernel<<<ewg(B * 2 * HW), dim3(256), 0, stream>>>(coords1, flow2, B, H, W);
    upsample_kernel<<<ewg(B * 64 * HW), dim3(256), 0, stream>>>(
        flow2, maskB, outF + (size_t)it * outStride, B, H, W);
  }
}
